// ShiftWindowAttentionBlock_66305705116267
// MI455X (gfx1250) — matrix operand order repaired and checked by reading, not compiled, not run on hardware
//
#include <hip/hip_runtime.h>
#include <hip/hip_bf16.h>

// ---------------------------------------------------------------------------
// Problem constants (from reference): B=32, M=8, P=256, D=512, H=4, A=64, WS=4
// ---------------------------------------------------------------------------
#define PP   256          // patches
#define DD   512          // model dim
#define HA   256          // H*A
#define SHF  2            // shift
#define BM   256          // B*M

typedef __attribute__((ext_vector_type(16))) __bf16 v16bf;
typedef __attribute__((ext_vector_type(8)))  __bf16 v8bf;
typedef __attribute__((ext_vector_type(8)))  float  v8f;
typedef __attribute__((ext_vector_type(4)))  float  v4f;

__device__ __forceinline__ v8f wmma_bf16(v16bf a, v16bf b, v8f c) {
  // D = A(16x32 bf16) * B(32x16 bf16) + C(16x16 f32)
  return __builtin_amdgcn_wmma_f32_16x16x32_bf16(false, a, false, b, (short)0, c,
                                                 false, false);
}

// A-matrix fragment (16x32 bf16) from a row-major [16 x rowStride] bf16 tile.
// Contiguous k order (same as load_b_frag): lane<16 holds row=lane, K in {0..15}; lane>=16 holds
// row=lane-16, K in {16..31}.
__device__ __forceinline__ v16bf load_a_frag(const __bf16* base, int rowStride,
                                             int lane, int kbase) {
  int row = lane & 15;
  int kh  = (lane >> 4) * 16;
  const __bf16* p = base + row * rowStride + kbase;
  v8bf lo = *(const v8bf*)(p + kh);
  v8bf hi = *(const v8bf*)(p + kh + 8);
  v16bf a;
#pragma unroll
  for (int i = 0; i < 8; ++i) { a[i] = lo[i]; a[i + 8] = hi[i]; }
  return a;
}

// B-matrix fragment (32x16 bf16) straight from a row-major weight W[N][K]
// (output-major => B[k][n] = W[n][k], so a lane's column is a contiguous row).
// Contiguous k order, as in A: lane n<16 -> col n, K=kbase..kbase+15; lane>=16 -> col n-16,
// K=kbase+16..kbase+31. 16 contiguous halves per lane.
__device__ __forceinline__ v16bf load_b_frag(const __bf16* w, int rowStride,
                                             int lane, int ntile, int kbase) {
  int n  = ntile * 16 + (lane & 15);
  int k0 = kbase + (lane >> 4) * 16;
  return *(const v16bf*)(w + (size_t)n * rowStride + k0);
}

__device__ __forceinline__ int zone_of(int ps) {   // shift-window mask zones
  return (ps >= 254) ? 2 : ((ps >= 252) ? 1 : 0);
}

// ---------------------------------------------------------------------------
// Kernel 0: fp32 -> bf16 weight conversion
// ---------------------------------------------------------------------------
__global__ void cvt_bf16_kernel(const float* __restrict__ src,
                                __bf16* __restrict__ dst, int n) {
  int i = blockIdx.x * blockDim.x + threadIdx.x;
  if (i < n) dst[i] = (__bf16)src[i];
}

// ---------------------------------------------------------------------------
// Kernel 1: fused shifted-window attention block (everything up to FFN output)
// One workgroup = one (b,m) image x one 16-token tile (4 windows).
// 512 threads = 16 wave32s. Writes bf16 activations (pre-merge) to ws.
// ---------------------------------------------------------------------------
__global__ __launch_bounds__(512)
void swin_fused_kernel(const float* __restrict__ x,
                       const __bf16* __restrict__ Wq,
                       const __bf16* __restrict__ Wk,
                       const __bf16* __restrict__ Wv,
                       const __bf16* __restrict__ aWw,
                       const __bf16* __restrict__ fWw,
                       const float* __restrict__ bq, const float* __restrict__ bk,
                       const float* __restrict__ bv, const float* __restrict__ ab,
                       const float* __restrict__ ag, const float* __restrict__ abeta,
                       const float* __restrict__ fb, const float* __restrict__ fg,
                       const float* __restrict__ fbeta,
                       __bf16* __restrict__ outbuf) {
  // LDS map (105 KB, heavy region reuse):
  //   [0      ) xf    16x512 f32  (shifted x; also the residual rows)
  //   [32768  ) xbf   16x512 bf16 (A for QKV)      -> reused as h1 bf16
  //   [49152  ) q,k,v 3x16x256 f32                 -> first 32KB reused as
  //                                                   zo / h1 f32 (in place)
  //   [98304  ) s     256 f32 scores
  //   [99328  ) zattn 16x256 bf16 (A for out-proj)
  //   y (FFN out, 16x512 f32) reuses xf region.
  __shared__ __align__(32) unsigned char smem[107520];
  float*  xf    = (float*)(smem + 0);
  __bf16* xbf   = (__bf16*)(smem + 32768);
  float*  qs    = (float*)(smem + 49152);
  float*  ks    = qs + 16 * HA;
  float*  vs    = ks + 16 * HA;
  float*  sc    = (float*)(smem + 98304);
  __bf16* zattn = (__bf16*)(smem + 99328);
  float*  zo    = qs;              // 16x512 f32, aliases q+k (dead by then)
  float*  h1f   = zo;              // LN+residual written in place
  __bf16* h1bf  = xbf;             // bf16 copy of h1 (xbf dead)
  float*  yy    = xf;              // FFN GEMM output (xf dead after residual)

  const int bm      = blockIdx.x >> 4;        // which (b,m) image
  const int rowbase = (blockIdx.x & 15) * 16; // shifted-coord tile base
  const float* xim  = x + (size_t)bm * PP * DD;
  __bf16* oim       = outbuf + (size_t)bm * PP * DD;

  const int t    = threadIdx.x;
  const int wave = t >> 5;
  const int lane = t & 31;

  // ---- Stage 0: load shifted rows (xs[p] = x[(p+SHF) % P]) ----------------
  {
    const int r  = t >> 5;             // 16 rows x 32 threads
    const int c0 = (t & 31) * 16;      // 16 cols per thread
    const int gr = (rowbase + r + SHF) & (PP - 1);
    const float* src = xim + (size_t)gr * DD + c0;
    float*  dxf = xf  + r * DD + c0;
    __bf16* dxb = xbf + r * DD + c0;
#pragma unroll
    for (int i = 0; i < 16; i += 4) {
      v4f vv = *(const v4f*)(src + i);
      *(v4f*)(dxf + i) = vv;
      dxb[i + 0] = (__bf16)vv[0]; dxb[i + 1] = (__bf16)vv[1];
      dxb[i + 2] = (__bf16)vv[2]; dxb[i + 3] = (__bf16)vv[3];
    }
  }
  __syncthreads();

  // ---- Stage 1: Q/K/V = Xs @ W^T + b   (16 x 512 x 256, one ntile/wave) ---
  // One A fragment feeds 3 WMMAs (Q,K,V streams).
  {
    v8f accq = {}, acck = {}, accv = {};
    const int ntile = wave;            // N=256 -> 16 tiles
#pragma unroll
    for (int kk = 0; kk < 16; ++kk) {
      const int kb = kk * 32;
      v16bf a  = load_a_frag(xbf, DD, lane, kb);
      accq = wmma_bf16(a, load_b_frag(Wq, DD, lane, ntile, kb), accq);
      acck = wmma_bf16(a, load_b_frag(Wk, DD, lane, ntile, kb), acck);
      accv = wmma_bf16(a, load_b_frag(Wv, DD, lane, ntile, kb), accv);
    }
    const int col = ntile * 16 + (lane & 15);
    const int rb  = (lane >> 4) * 8;
    const float bqv = bq[col], bkv = bk[col], bvv = bv[col];
#pragma unroll
    for (int e = 0; e < 8; ++e) {
      qs[(rb + e) * HA + col] = accq[e] + bqv;
      ks[(rb + e) * HA + col] = acck[e] + bkv;
      vs[(rb + e) * HA + col] = accv[e] + bvv;
    }
  }
  __syncthreads();

  // ---- Stage 2: windowed 4x4 attention -----------------------------------
  if (t < 256) {                       // one score per thread: (w4,h,i,j)
    const int w4 = t >> 6, h = (t >> 4) & 3, i = (t >> 2) & 3, j = t & 3;
    const int ri = w4 * 4 + i, rj = w4 * 4 + j;
    const float* qp = qs + ri * HA + h * 64;
    const float* kp = ks + rj * HA + h * 64;
    float s = 0.f;
#pragma unroll
    for (int a = 0; a < 64; ++a) s += qp[a] * kp[a];
    s *= 0.125f;                       // 1/sqrt(64)
    if (zone_of(rowbase + ri) != zone_of(rowbase + rj)) s -= 100.0f;
    sc[t] = s;
  }
  __syncthreads();
  if (t < 64) {                        // softmax over j for each (w4,h,i)
    float* sp = sc + t * 4;
    float m = fmaxf(fmaxf(sp[0], sp[1]), fmaxf(sp[2], sp[3]));
    float e0 = expf(sp[0] - m), e1 = expf(sp[1] - m);
    float e2 = expf(sp[2] - m), e3 = expf(sp[3] - m);
    float inv = 1.0f / (e0 + e1 + e2 + e3);
    sp[0] = e0 * inv; sp[1] = e1 * inv; sp[2] = e2 * inv; sp[3] = e3 * inv;
  }
  __syncthreads();
  {                                    // z = p @ v  -> bf16 tile [16 x 256]
    const int r  = wave;               // row
    const int w4 = r >> 2, i = r & 3;
    const int c0 = lane * 8;           // 8 contiguous cols, same head
    const int h  = lane >> 3;
    const float* pw = sc + (w4 * 64 + h * 16 + i * 4);
    const float p0 = pw[0], p1 = pw[1], p2 = pw[2], p3 = pw[3];
    const float* v0 = vs + (w4 * 4 + 0) * HA + c0;
    const float* v1 = vs + (w4 * 4 + 1) * HA + c0;
    const float* v2 = vs + (w4 * 4 + 2) * HA + c0;
    const float* v3 = vs + (w4 * 4 + 3) * HA + c0;
    __bf16* zr = zattn + r * HA + c0;
#pragma unroll
    for (int c = 0; c < 8; ++c)
      zr[c] = (__bf16)(p0 * v0[c] + p1 * v1[c] + p2 * v2[c] + p3 * v3[c]);
  }
  __syncthreads();

  // ---- Stage 3: out-proj zo = z @ aW^T + ab  (16 x 256 x 512) -------------
  // Both N-tile accumulators live in the k-loop: each A fragment loaded once.
  {
    const int nt0 = wave * 2, nt1 = wave * 2 + 1;  // N=512 -> 32 tiles
    v8f acc0 = {}, acc1 = {};
#pragma unroll
    for (int kk = 0; kk < 8; ++kk) {
      const int kb = kk * 32;
      v16bf a = load_a_frag(zattn, HA, lane, kb);
      acc0 = wmma_bf16(a, load_b_frag(aWw, HA, lane, nt0, kb), acc0);
      acc1 = wmma_bf16(a, load_b_frag(aWw, HA, lane, nt1, kb), acc1);
    }
    const int c0 = nt0 * 16 + (lane & 15);
    const int c1 = nt1 * 16 + (lane & 15);
    const int rb = (lane >> 4) * 8;
    const float b0 = ab[c0], b1 = ab[c1];
#pragma unroll
    for (int e = 0; e < 8; ++e) {
      zo[(rb + e) * DD + c0] = acc0[e] + b0;
      zo[(rb + e) * DD + c1] = acc1[e] + b1;
    }
  }
  __syncthreads();

  // ---- Stage 4: LN + un-shift residual + ReLU (wave r owns row r) ---------
  {
    const int r  = wave;
    float* zrow  = zo + r * DD;
    const int c0 = lane * 16;
    float vals[16];
    float sum = 0.f, sq = 0.f;
#pragma unroll
    for (int i = 0; i < 16; ++i) {
      float vv = zrow[c0 + i]; vals[i] = vv; sum += vv; sq += vv * vv;
    }
#pragma unroll
    for (int off = 16; off > 0; off >>= 1) {
      sum += __shfl_xor(sum, off); sq += __shfl_xor(sq, off);
    }
    const float mu = sum * (1.0f / DD);
    const float rs = rsqrtf(sq * (1.0f / DD) - mu * mu + 1e-5f);
#pragma unroll
    for (int i = 0; i < 16; ++i) {
      const int c = c0 + i;
      float ln = (vals[i] - mu) * rs * ag[c] + abeta[c];
      float h  = fmaxf(xf[r * DD + c] + ln, 0.f);   // residual = shifted row
      zrow[c0 + i]     = h;                          // h1f (in place)
      h1bf[r * DD + c] = (__bf16)h;
    }
  }
  __syncthreads();

  // ---- Stage 5: FFN y = h1 @ fW^T + fb  (16 x 512 x 512) ------------------
  {
    const int nt0 = wave * 2, nt1 = wave * 2 + 1;
    v8f acc0 = {}, acc1 = {};
#pragma unroll
    for (int kk = 0; kk < 16; ++kk) {
      const int kb = kk * 32;
      v16bf a = load_a_frag(h1bf, DD, lane, kb);
      acc0 = wmma_bf16(a, load_b_frag(fWw, DD, lane, nt0, kb), acc0);
      acc1 = wmma_bf16(a, load_b_frag(fWw, DD, lane, nt1, kb), acc1);
    }
    const int c0 = nt0 * 16 + (lane & 15);
    const int c1 = nt1 * 16 + (lane & 15);
    const int rb = (lane >> 4) * 8;
    const float b0 = fb[c0], b1 = fb[c1];
#pragma unroll
    for (int e = 0; e < 8; ++e) {
      yy[(rb + e) * DD + c0] = acc0[e] + b0;
      yy[(rb + e) * DD + c1] = acc1[e] + b1;
    }
  }
  __syncthreads();

  // ---- Stage 6: out = relu(h1 + LN(y)) -> bf16 global (un-shifted row) ----
  {
    const int r  = wave;
    const int gr = (rowbase + r + SHF) & (PP - 1);
    float* yrow  = yy + r * DD;
    const int c0 = lane * 16;
    float vals[16];
    float sum = 0.f, sq = 0.f;
#pragma unroll
    for (int i = 0; i < 16; ++i) {
      float vv = yrow[c0 + i]; vals[i] = vv; sum += vv; sq += vv * vv;
    }
#pragma unroll
    for (int off = 16; off > 0; off >>= 1) {
      sum += __shfl_xor(sum, off); sq += __shfl_xor(sq, off);
    }
    const float mu = sum * (1.0f / DD);
    const float rs = rsqrtf(sq * (1.0f / DD) - mu * mu + 1e-5f);
    __bf16* dst = oim + (size_t)gr * DD;
#pragma unroll
    for (int i = 0; i < 16; ++i) {
      const int c = c0 + i;
      float ln = (vals[i] - mu) * rs * fg[c] + fbeta[c];
      dst[c] = (__bf16)fmaxf(h1f[r * DD + c] + ln, 0.f);
    }
  }
}

// ---------------------------------------------------------------------------
// Kernel 2: patch merging + GEMM + LN + ReLU.
// merged[b,m,i,:] = concat(out[2i], out[2i+1]) == contiguous 1024 bf16 row.
// One workgroup = 16 merged rows x full N=512. 512 threads = 16 waves.
// ---------------------------------------------------------------------------
__global__ __launch_bounds__(512)
void merge_kernel(const __bf16* __restrict__ outbuf,
                  const __bf16* __restrict__ dWw,
                  const float* __restrict__ db, const float* __restrict__ dg,
                  const float* __restrict__ dbeta,
                  float* __restrict__ out) {
  __shared__ __align__(32) unsigned char smem[65536];
  __bf16* As = (__bf16*)(smem + 0);        // 16 x 1024 bf16 (32 KB)
  float*  yy = (float*)(smem + 32768);     // 16 x 512 f32  (32 KB)

  const int bm    = blockIdx.x >> 3;       // (b,m)
  const int mtile = blockIdx.x & 7;        // 128 merged rows / 16
  const __bf16* src = outbuf + (size_t)bm * PP * DD + (size_t)mtile * 16 * 1024;

  const int t    = threadIdx.x;
  const int wave = t >> 5;
  const int lane = t & 31;

  // load A tile: 16384 bf16, 32 per thread (free "merge": rows are contiguous)
  {
    const __bf16* s = src + t * 32;
    __bf16* d = As + t * 32;
#pragma unroll
    for (int i = 0; i < 32; i += 8) *(v8bf*)(d + i) = *(const v8bf*)(s + i);
  }
  __syncthreads();

  // GEMM: 16 x 1024 x 512, two N-tiles per wave, A fragment loaded once
  {
    const int nt0 = wave * 2, nt1 = wave * 2 + 1;
    v8f acc0 = {}, acc1 = {};
#pragma unroll
    for (int kk = 0; kk < 32; ++kk) {
      const int kb = kk * 32;
      v16bf a = load_a_frag(As, 1024, lane, kb);
      acc0 = wmma_bf16(a, load_b_frag(dWw, 1024, lane, nt0, kb), acc0);
      acc1 = wmma_bf16(a, load_b_frag(dWw, 1024, lane, nt1, kb), acc1);
    }
    const int c0 = nt0 * 16 + (lane & 15);
    const int c1 = nt1 * 16 + (lane & 15);
    const int rb = (lane >> 4) * 8;
    const float b0 = db[c0], b1 = db[c1];
#pragma unroll
    for (int e = 0; e < 8; ++e) {
      yy[(rb + e) * DD + c0] = acc0[e] + b0;
      yy[(rb + e) * DD + c1] = acc1[e] + b1;
    }
  }
  __syncthreads();

  // LN + ReLU, f32 store (wave r owns row r)
  {
    const int r  = wave;
    float* yrow  = yy + r * DD;
    const int c0 = lane * 16;
    float vals[16];
    float sum = 0.f, sq = 0.f;
#pragma unroll
    for (int i = 0; i < 16; ++i) {
      float vv = yrow[c0 + i]; vals[i] = vv; sum += vv; sq += vv * vv;
    }
#pragma unroll
    for (int off = 16; off > 0; off >>= 1) {
      sum += __shfl_xor(sum, off); sq += __shfl_xor(sq, off);
    }
    const float mu = sum * (1.0f / DD);
    const float rs = rsqrtf(sq * (1.0f / DD) - mu * mu + 1e-5f);
    const int mr = mtile * 16 + r;
    float* dst = out + (size_t)bm * 128 * DD + (size_t)mr * DD;
#pragma unroll
    for (int i = 0; i < 16; ++i) {
      const int c = c0 + i;
      dst[c] = fmaxf((vals[i] - mu) * rs * dg[c] + dbeta[c], 0.f);
    }
  }
}

// ---------------------------------------------------------------------------
// Launch. ws layout:
//   [0       ) Wq bf16   256KB      [786432 ) aW bf16 256KB
//   [262144  ) Wk bf16   256KB      [1048576) fW bf16 512KB
//   [524288  ) Wv bf16   256KB      [1572864) dW bf16 1MB
//   [2621440 ) activation buffer bf16 [B,M,P,D] = 64MB
// ---------------------------------------------------------------------------
extern "C" void kernel_launch(void* const* d_in, const int* in_sizes, int n_in,
                              void* d_out, int out_size, void* d_ws, size_t ws_size,
                              hipStream_t stream) {
  const float* x     = (const float*)d_in[0];
  const float* Wq    = (const float*)d_in[1];
  const float* bq    = (const float*)d_in[2];
  const float* Wk    = (const float*)d_in[3];
  const float* bk    = (const float*)d_in[4];
  const float* Wv    = (const float*)d_in[5];
  const float* bv    = (const float*)d_in[6];
  const float* aW    = (const float*)d_in[7];
  const float* ab    = (const float*)d_in[8];
  const float* ag    = (const float*)d_in[9];
  const float* abeta = (const float*)d_in[10];
  const float* fW    = (const float*)d_in[11];
  const float* fb    = (const float*)d_in[12];
  const float* fg    = (const float*)d_in[13];
  const float* fbeta = (const float*)d_in[14];
  const float* dW    = (const float*)d_in[15];
  const float* db    = (const float*)d_in[16];
  const float* dg    = (const float*)d_in[17];
  const float* dbeta = (const float*)d_in[18];

  unsigned char* ws = (unsigned char*)d_ws;
  __bf16* wqb  = (__bf16*)(ws + 0);
  __bf16* wkb  = (__bf16*)(ws + 262144);
  __bf16* wvb  = (__bf16*)(ws + 524288);
  __bf16* awb  = (__bf16*)(ws + 786432);
  __bf16* fwb  = (__bf16*)(ws + 1048576);
  __bf16* dwb  = (__bf16*)(ws + 1572864);
  __bf16* obuf = (__bf16*)(ws + 2621440);

  cvt_bf16_kernel<<<(131072 + 255) / 256, 256, 0, stream>>>(Wq, wqb, 131072);
  cvt_bf16_kernel<<<(131072 + 255) / 256, 256, 0, stream>>>(Wk, wkb, 131072);
  cvt_bf16_kernel<<<(131072 + 255) / 256, 256, 0, stream>>>(Wv, wvb, 131072);
  cvt_bf16_kernel<<<(131072 + 255) / 256, 256, 0, stream>>>(aW, awb, 131072);
  cvt_bf16_kernel<<<(262144 + 255) / 256, 256, 0, stream>>>(fW, fwb, 262144);
  cvt_bf16_kernel<<<(524288 + 255) / 256, 256, 0, stream>>>(dW, dwb, 524288);

  swin_fused_kernel<<<BM * 16, 512, 0, stream>>>(
      x, wqb, wkb, wvb, awb, fwb, bq, bk, bv, ab, ag, abeta, fb, fg, fbeta, obuf);

  merge_kernel<<<BM * 8, 512, 0, stream>>>(obuf, dwb, db, dg, dbeta,
                                           (float*)d_out);
}
